// BigramSeg_50620484550792
// MI455X (gfx1250) — matrix operand order repaired and checked by reading, not compiled, not run on hardware
//
#include <hip/hip_runtime.h>
#include <hip/hip_bf16.h>
#include <math.h>

// Problem constants (B=256, T=512, D=1024, H=2048, O=32)
#define BATCH 256
#define TLEN  512
#define DIN   1024
#define HID   2048
#define OUTC  32

typedef __bf16 v16bf __attribute__((ext_vector_type(16)));
typedef __bf16 v8bf  __attribute__((ext_vector_type(8)));
typedef float  v8f   __attribute__((ext_vector_type(8)));

static __device__ __forceinline__ unsigned short bfbits(float f) {
    __bf16 b = (__bf16)f;                      // RNE f32->bf16 convert
    return __builtin_bit_cast(unsigned short, b);
}

// gfx1250 hardware tanh (TRANS op, co-executes with WMMA)
static __device__ __forceinline__ float fast_tanh(float x) {
#if __has_builtin(__builtin_amdgcn_tanhf)
    return __builtin_amdgcn_tanhf(x);
#elif __has_builtin(__builtin_amdgcn_tanh_f32)
    return __builtin_amdgcn_tanh_f32(x);
#else
    float r;
    asm("v_tanh_f32 %0, %1" : "=v"(r) : "v"(x));
    return r;
#endif
}

// A-fragment (16x32 bf16, M = lane&15): lanes 0-15 need K {0..7}U{16..23},
// lanes 16-31 need K {8..15}U{24..31}. Two 16B chunks from a row-major K line.
static __device__ __forceinline__ v16bf fragA(const unsigned short* rowp, int kbase, int lane) {
    const int koff = (lane & 16) >> 1;   // 0 or 8
    v8bf lo, hi;
    __builtin_memcpy(&lo, rowp + kbase + koff,      16);
    __builtin_memcpy(&hi, rowp + kbase + koff + 16, 16);
    return __builtin_shufflevector(lo, hi, 0,1,2,3,4,5,6,7,8,9,10,11,12,13,14,15);
}

// B-fragment (32x16 bf16, N = lane&15) built from transposed [N][K] storage:
// lanes 0-15 hold K {0..7}U{16..23}, lanes 16-31 hold K {8..15}U{24..31} -> two 16B reads.
static __device__ __forceinline__ v16bf fragB(const unsigned short* rowp, int kbase, int lane) {
    const int koff = (lane & 16) >> 1;   // 0 or 8
    v8bf lo, hi;
    __builtin_memcpy(&lo, rowp + kbase + koff,      16);
    __builtin_memcpy(&hi, rowp + kbase + koff + 16, 16);
    return __builtin_shufflevector(lo, hi, 0,1,2,3,4,5,6,7,8,9,10,11,12,13,14,15);
}

static __device__ __forceinline__ v8f wmma_bf16(v16bf a, v16bf b, v8f c) {
    return __builtin_amdgcn_wmma_f32_16x16x32_bf16(false, a, false, b, (short)0, c, false, false);
}

// ---- prep kernels -----------------------------------------------------------
// W1top (D x H row-major f32) -> W1T bf16 [H][D]
__global__ void conv_w1t_kernel(const float* __restrict__ W1, unsigned short* __restrict__ w1t) {
    int idx = blockIdx.x * 256 + threadIdx.x;          // 2048*1024 elems
    if (idx >= HID * DIN) return;
    int n = idx & (HID - 1);
    int k = idx >> 11;
    w1t[(size_t)n * DIN + k] = bfbits(W1[(size_t)k * HID + n]);
}

// W2 (H x O row-major f32) -> W2T bf16 [O][H]
__global__ void conv_w2t_kernel(const float* __restrict__ W2, unsigned short* __restrict__ w2t) {
    int idx = blockIdx.x * 256 + threadIdx.x;          // 2048*32 elems
    if (idx >= HID * OUTC) return;
    int n = idx & (OUTC - 1);
    int k = idx >> 5;
    w2t[(size_t)n * HID + k] = bfbits(W2[(size_t)k * OUTC + n]);
}

// zero pred ping-pong (2x256) + global barrier counter (1)
__global__ void init_state_kernel(int* __restrict__ p) {
    int idx = blockIdx.x * 256 + threadIdx.x;
    if (idx < 2 * BATCH + 1) p[idx] = 0;
}

// ---- Phase 1: pre[t][b][h] = x[b][t][:] @ W1top + b1 ------------------------
// M = B*T = 131072, N = H = 2048, K = D = 1024. BM=BN=128, BK=64, 256 thr.
__global__ __launch_bounds__(256) void gemm_pre_kernel(
        const float* __restrict__ x, const unsigned short* __restrict__ w1t,
        const float* __restrict__ b1, float* __restrict__ pre) {
    __shared__ unsigned short sA[128][64];   // 16 KB
    __shared__ unsigned short sB[128][64];   // 16 KB

    const int tid  = threadIdx.x;
    const int lane = tid & 31;
    const int wave = tid >> 5;          // 8 waves
    const int wm   = wave & 3;          // 4 in M -> 32 rows each
    const int wn   = wave >> 2;         // 2 in N -> 64 cols each
    const int mTile = blockIdx.x >> 4;  // 1024 M tiles
    const int nTile = blockIdx.x & 15;  // 16 N tiles
    const int m0 = mTile * 128;
    const int n0 = nTile * 128;

    const v8f vz = {0.f,0.f,0.f,0.f,0.f,0.f,0.f,0.f};
    v8f acc[2][4];
#pragma unroll
    for (int i = 0; i < 2; ++i)
#pragma unroll
        for (int j = 0; j < 4; ++j) acc[i][j] = vz;

    const int arow = tid >> 4;           // 0..15
    const int acol = (tid & 15) * 4;     // 0..60
    const int brow = tid >> 1;           // 0..127
    const int bcol = (tid & 1) * 32;     // 0 / 32
    const unsigned sBbase = (unsigned)(uintptr_t)&sB[0][0];  // LDS byte offset

    for (int kk = 0; kk < DIN; kk += 64) {
        // W1T tile: bf16 global -> LDS directly via CDNA5 async-to-LDS DMA path
#pragma unroll
        for (int cc = 0; cc < 4; ++cc) {
            const int col = bcol + cc * 8;
            const unsigned long long gp =
                (unsigned long long)(uintptr_t)(w1t + (size_t)(n0 + brow) * DIN + kk + col);
            const unsigned lp = sBbase + (unsigned)(brow * 64 + col) * 2u;
            asm volatile("global_load_async_to_lds_b128 %0, %1, off"
                         :: "v"(lp), "v"(gp) : "memory");
        }
        // x tile: f32 global -> bf16 LDS (coalesced float4 loads + cvt_pk_bf16)
#pragma unroll
        for (int rr = 0; rr < 8; ++rr) {
            const int row = arow + rr * 16;
            const float4 v = *(const float4*)(x + (size_t)(m0 + row) * DIN + kk + acol);
            ushort4 pk;
            pk.x = bfbits(v.x); pk.y = bfbits(v.y); pk.z = bfbits(v.z); pk.w = bfbits(v.w);
            *(ushort4*)&sA[row][acol] = pk;
        }
        asm volatile("s_wait_asynccnt 0x0" ::: "memory");
        __syncthreads();

#pragma unroll
        for (int ks = 0; ks < 64; ks += 32) {
            v16bf af[2], bf[4];
#pragma unroll
            for (int mi = 0; mi < 2; ++mi)
                af[mi] = fragA(&sA[wm * 32 + mi * 16 + (lane & 15)][0], ks, lane);
#pragma unroll
            for (int ni = 0; ni < 4; ++ni)
                bf[ni] = fragB(&sB[wn * 64 + ni * 16 + (lane & 15)][0], ks, lane);
#pragma unroll
            for (int mi = 0; mi < 2; ++mi)
#pragma unroll
                for (int ni = 0; ni < 4; ++ni)
                    acc[mi][ni] = wmma_bf16(af[mi], bf[ni], acc[mi][ni]);
        }
        __syncthreads();
    }

    // Epilogue: +b1, scatter rows m=b*T+t into time-major pre[t*B+b][n]
#pragma unroll
    for (int mi = 0; mi < 2; ++mi) {
#pragma unroll
        for (int ni = 0; ni < 4; ++ni) {
            const int gn = n0 + wn * 64 + ni * 16 + (lane & 15);
            const float bias = b1[gn];
            const int mbase = m0 + wm * 32 + mi * 16 + ((lane >> 4) << 3);
#pragma unroll
            for (int r = 0; r < 8; ++r) {
                const int m = mbase + r;
                const int b = m >> 9;        // /T
                const int t = m & (TLEN - 1);
                pre[(size_t)(t * BATCH + b) * HID + gn] = acc[mi][ni][r] + bias;
            }
        }
    }
}

// ---- Phase 2: persistent sequential decoder ---------------------------------
// Single launch, grid 16 (16 batch rows each), 128 threads = 4 waves splitting
// K=2048. Cross-WG step ordering via monotonic device-scope atomic barrier.
__global__ __launch_bounds__(128) void decode_kernel(
        const float* __restrict__ pre, const float* __restrict__ W1,
        const unsigned short* __restrict__ w2t, const float* __restrict__ b2,
        int* __restrict__ predbuf, int* __restrict__ bar,
        float* __restrict__ logits, long long* __restrict__ preds64) {
    __shared__ unsigned short sH[16][1024];   // half of h, bf16 (32 KB)
    __shared__ float sC[4][16][32];           // per-wave partial C
    __shared__ float sL[16][32];              // final logits
    __shared__ int   sPred[16];

    const int tid  = threadIdx.x;
    const int lane = tid & 31;
    const int wave = tid >> 5;
    const int b0   = blockIdx.x * 16;

    for (int t = 0; t < TLEN; ++t) {
        const int* predIn  = predbuf + (t & 1) * BATCH;
        int*       predOut = predbuf + ((t + 1) & 1) * BATCH;
        const float* preT  = pre + (size_t)t * BATCH * HID;

        if (tid < 16) sPred[tid] = predIn[b0 + tid] & (OUTC - 1);
        __syncthreads();

        // prefetch a slice of next step's pre slab (global_prefetch_b8)
        if (t + 1 < TLEN) {
            const char* nxt = (const char*)(preT + (size_t)BATCH * HID + (size_t)b0 * HID);
#pragma unroll
            for (int p = 0; p < 4; ++p)
                __builtin_prefetch(nxt + (size_t)tid * 1024 + p * 256, 0, 1);
        }

        const v8f vz = {0.f,0.f,0.f,0.f,0.f,0.f,0.f,0.f};
        v8f acc0 = vz, acc1 = vz;

        for (int half = 0; half < 2; ++half) {
            // h = tanh(pre + W1_feedback[pred]) for K slice [half*1024, +1024)
#pragma unroll 4
            for (int i = 0; i < 32; ++i) {
                const int e4  = tid + i * 128;
                const int row = e4 >> 8;           // 256 float4 per row
                const int col = (e4 & 255) * 4;
                const int gc  = half * 1024 + col;
                const float4 pv = *(const float4*)(preT + (size_t)(b0 + row) * HID + gc);
                const float4 wv = *(const float4*)(W1 + (size_t)(DIN + sPred[row]) * HID + gc);
                ushort4 pk;
                pk.x = bfbits(fast_tanh(pv.x + wv.x));
                pk.y = bfbits(fast_tanh(pv.y + wv.y));
                pk.z = bfbits(fast_tanh(pv.z + wv.z));
                pk.w = bfbits(fast_tanh(pv.w + wv.w));
                *(ushort4*)&sH[row][col] = pk;
            }
            __syncthreads();

            const int kw = wave * 256;             // this wave's K window in half
#pragma unroll
            for (int ks = 0; ks < 8; ++ks) {
                const int k  = kw + ks * 32;
                const int gk = half * 1024 + k;
                const v16bf a  = fragA(&sH[lane & 15][0], k, lane);
                const v16bf bA = fragB(w2t + (size_t)(lane & 15) * HID, gk, lane);
                const v16bf bB = fragB(w2t + (size_t)(16 + (lane & 15)) * HID, gk, lane);
                acc0 = wmma_bf16(a, bA, acc0);
                acc1 = wmma_bf16(a, bB, acc1);
            }
            __syncthreads();                       // before overwriting sH
        }

        // dump per-wave partials
        {
            const int m = (lane >> 4) << 3;
            const int n = lane & 15;
#pragma unroll
            for (int r = 0; r < 8; ++r) {
                sC[wave][m + r][n]      = acc0[r];
                sC[wave][m + r][n + 16] = acc1[r];
            }
        }
        __syncthreads();

        // reduce 4 waves, +b2, emit logits
#pragma unroll
        for (int j = 0; j < 4; ++j) {
            const int idx = tid + j * 128;
            const int m = idx >> 5, n = idx & 31;
            const float v = sC[0][m][n] + sC[1][m][n] + sC[2][m][n] + sC[3][m][n] + b2[n];
            sL[m][n] = v;
            logits[((size_t)(b0 + m) * TLEN + t) * OUTC + n] = v;
        }
        __syncthreads();

        // argmax (first max wins, matching jnp.argmax) + feedback + int64 preds
        if (tid < 16) {
            float best = sL[tid][0];
            int bi = 0;
#pragma unroll
            for (int n = 1; n < OUTC; ++n) {
                const float v = sL[tid][n];
                if (v > best) { best = v; bi = n; }
            }
            predOut[b0 + tid] = bi;
            preds64[(size_t)(b0 + tid) * TLEN + t] = (long long)bi;
        }
        __syncthreads();

        // grid-wide barrier: all 16 WGs must finish step t before step t+1
        if (tid == 0) {
            __hip_atomic_fetch_add(bar, 1, __ATOMIC_RELEASE, __HIP_MEMORY_SCOPE_AGENT);
            const int target = 16 * (t + 1);
            while (__hip_atomic_load(bar, __ATOMIC_ACQUIRE, __HIP_MEMORY_SCOPE_AGENT) < target)
                __builtin_amdgcn_s_sleep(1);
        }
        __syncthreads();
    }
}

extern "C" void kernel_launch(void* const* d_in, const int* in_sizes, int n_in,
                              void* d_out, int out_size, void* d_ws, size_t ws_size,
                              hipStream_t stream) {
    const float* x  = (const float*)d_in[0];   // [256,512,1024]
    const float* W1 = (const float*)d_in[1];   // [1056,2048]
    const float* b1 = (const float*)d_in[2];   // [2048]
    const float* W2 = (const float*)d_in[3];   // [2048,32]
    const float* b2 = (const float*)d_in[4];   // [32]

    char* ws = (char*)d_ws;
    const size_t preBytes = (size_t)TLEN * BATCH * HID * sizeof(float);   // 1 GiB
    const size_t w1tBytes = (size_t)HID * DIN * sizeof(unsigned short);   // 4 MiB
    const size_t w2tBytes = (size_t)OUTC * HID * sizeof(unsigned short);  // 128 KiB
    float*          pre  = (float*)ws;
    unsigned short* w1t  = (unsigned short*)(ws + preBytes);
    unsigned short* w2t  = (unsigned short*)(ws + preBytes + w1tBytes);
    int*            st   = (int*)(ws + preBytes + w1tBytes + w2tBytes);   // preds[512] + bar[1]
    int*            pred = st;
    int*            bar  = st + 2 * BATCH;

    float*     logits  = (float*)d_out;                                   // [256,512,32] f32
    long long* preds64 = (long long*)((char*)d_out +
                         (size_t)BATCH * TLEN * OUTC * sizeof(float));    // [256,512] i64

    // prep
    conv_w1t_kernel<<<(HID * DIN) / 256, 256, 0, stream>>>(W1, w1t);
    conv_w2t_kernel<<<(HID * OUTC) / 256, 256, 0, stream>>>(W2, w2t);
    init_state_kernel<<<3, 256, 0, stream>>>(st);

    // phase 1: big WMMA GEMM (131072 x 2048 x 1024)
    gemm_pre_kernel<<<(BATCH * TLEN / 128) * (HID / 128), 256, 0, stream>>>(x, w1t, b1, pre);

    // phase 2: one persistent launch runs all 512 sequential steps
    decode_kernel<<<BATCH / 16, 128, 0, stream>>>(pre, W1, w2t, b2, pred, bar,
                                                  logits, preds64);
}
